// EdgeDecoder_81415400063670
// MI455X (gfx1250) — matrix operand order repaired and checked by reading, not compiled, not run on hardware
//
#include <hip/hip_runtime.h>
#include <hip/hip_bf16.h>

typedef __attribute__((ext_vector_type(16))) __bf16 v16bf;
typedef __attribute__((ext_vector_type(8)))  float  v8f;

union BFrag { uint4 u[2]; v16bf v; };

// Single-instruction bf16 pack: dst[15:0]=bf16(x), dst[31:16]=bf16(y) (RNE)
__device__ __forceinline__ unsigned pack2_bf16(float x, float y) {
  unsigned r;
  asm("v_cvt_pk_bf16_f32 %0, %1, %2" : "=v"(r) : "v"(x), "v"(y));
  return r;
}

__device__ __forceinline__ unsigned short bf16_1(float x) {
  return (unsigned short)(pack2_bf16(x, 0.0f) & 0xFFFFu);
}

#define WAVES 8   // waves per block (256 threads, wave32)
#define TPP   2   // passes per wave; each pass = 2 tiles (32 edges)

// Build one 16x32 bf16 A-fragment for K-chunk kc directly from global gathers.
// Lane (m,half): u[0] = z[m][kb+half*8 .. +7], u[1] = z[m][kb+half*8+16 .. +23]
__device__ __forceinline__ BFrag build_a(const float4* __restrict__ s,
                                         const float4* __restrict__ d,
                                         int kc, int half) {
  const int idx = kc * 8 + half * 2;   // float4 index of kb+half*8
  float4 xa = s[idx],     ya = d[idx];
  float4 xb = s[idx + 1], yb = d[idx + 1];
  float4 xc = s[idx + 4], yc = d[idx + 4];
  float4 xd = s[idx + 5], yd = d[idx + 5];
  BFrag f;
  f.u[0].x = pack2_bf16(xa.x * ya.x, xa.y * ya.y);
  f.u[0].y = pack2_bf16(xa.z * ya.z, xa.w * ya.w);
  f.u[0].z = pack2_bf16(xb.x * yb.x, xb.y * yb.y);
  f.u[0].w = pack2_bf16(xb.z * yb.z, xb.w * yb.w);
  f.u[1].x = pack2_bf16(xc.x * yc.x, xc.y * yc.y);
  f.u[1].y = pack2_bf16(xc.z * yc.z, xc.w * yc.w);
  f.u[1].z = pack2_bf16(xd.x * yd.x, xd.y * yd.y);
  f.u[1].w = pack2_bf16(xd.z * yd.z, xd.w * yd.w);
  return f;
}

__global__ __launch_bounds__(256) void edge_decoder_kernel(
    const float* __restrict__ src, const float* __restrict__ dst,
    const long long* __restrict__ eli, const float* __restrict__ W1,
    const float* __restrict__ b1, const float* __restrict__ W2,
    const float* __restrict__ b2, float* __restrict__ out, int n_edges) {

  __shared__ unsigned short w1t[128 * 128];   // W1 transposed [n][k], bf16 (32 KB)

  const int tid  = threadIdx.x;
  const int wave = tid >> 5;
  const int lane = tid & 31;
  const int m    = lane & 15;   // row in edge tile / column in N-tile
  const int half = lane >> 4;

  // ---- stage W1 transposed+bf16 into LDS (coalesced read, one-time) ----
  for (int idx = tid; idx < 128 * 128; idx += 256) {
    int k = idx >> 7, n = idx & 127;
    w1t[n * 128 + k] = bf16_1(W1[idx]);
  }
  __syncthreads();

  // per-lane epilogue constants
  float w2loc[8], b1loc[8];
#pragma unroll
  for (int nt = 0; nt < 8; ++nt) {
    w2loc[nt] = W2[nt * 16 + m];
    b1loc[nt] = b1[nt * 16 + m];
  }
  const float b2v = b2[0];

  const int total_tiles = (n_edges + 15) >> 4;
  const float4* src4 = (const float4*)src;
  const float4* dst4 = (const float4*)dst;
  const unsigned short* w1row = &w1t[m * 128 + half * 8];  // + nt*2048 + kc*32

  for (int p = 0; p < TPP; ++p) {
    const int tile0 = ((blockIdx.x * WAVES + wave) * TPP + p) * 2;
    if (tile0 >= total_tiles) break;   // uniform per wave -> EXEC all-1s
    const int tbase = tile0 << 4;      // 32 edges: tbase .. tbase+31

    int e0 = tbase + m;
    int e1 = tbase + 16 + m;
    int c0 = e0 < n_edges ? e0 : n_edges - 1;
    int c1 = e1 < n_edges ? e1 : n_edges - 1;
    const long long row0 = eli[c0], col0 = eli[(size_t)n_edges + c0];
    const long long row1 = eli[c1], col1 = eli[(size_t)n_edges + c1];

    const float4* s0 = src4 + (size_t)row0 * 32;
    const float4* d0 = dst4 + (size_t)col0 * 32;
    const float4* s1 = src4 + (size_t)row1 * 32;
    const float4* d1 = dst4 + (size_t)col1 * 32;

    v8f acc0[8] = {};
    v8f acc1[8] = {};

    // B fragment double-buffer: prime with (kc=0, nt=0)
    BFrag bcur;
    {
      const uint4* bp = (const uint4*)w1row;
      bcur.u[0] = bp[0];
      bcur.u[1] = bp[2];
    }

#pragma unroll
    for (int kc = 0; kc < 4; ++kc) {
      BFrag a0 = build_a(s0, d0, kc, half);
      BFrag a1 = build_a(s1, d1, kc, half);
#pragma unroll
      for (int nt = 0; nt < 8; ++nt) {
        // prefetch next B fragment (possibly crossing the kc boundary)
        BFrag bnxt = bcur;
        if (!(kc == 3 && nt == 7)) {
          const int knext = (nt < 7) ? kc : kc + 1;
          const int nnext = (nt < 7) ? nt + 1 : 0;
          const uint4* bp =
              (const uint4*)(w1row + nnext * 16 * 128 + knext * 32);
          bnxt.u[0] = bp[0];
          bnxt.u[1] = bp[2];
        }
        acc0[nt] = __builtin_amdgcn_wmma_f32_16x16x32_bf16(
            false, a0.v, false, bcur.v, (short)0, acc0[nt], false, false);
        acc1[nt] = __builtin_amdgcn_wmma_f32_16x16x32_bf16(
            false, a1.v, false, bcur.v, (short)0, acc1[nt], false, false);
        bcur = bnxt;
      }
    }

    // epilogue: h = relu(acc + b1); s += h * W2; 16-lane tree reduce; store
    float sa[8], sb[8];
#pragma unroll
    for (int r = 0; r < 8; ++r) { sa[r] = 0.0f; sb[r] = 0.0f; }
#pragma unroll
    for (int nt = 0; nt < 8; ++nt) {
      float bias = b1loc[nt];
      float w2v  = w2loc[nt];
#pragma unroll
      for (int r = 0; r < 8; ++r) {
        float h0 = acc0[nt][r] + bias;
        float h1 = acc1[nt][r] + bias;
        h0 = h0 > 0.0f ? h0 : 0.0f;
        h1 = h1 > 0.0f ? h1 : 0.0f;
        sa[r] += h0 * w2v;
        sb[r] += h1 * w2v;
      }
    }
#pragma unroll
    for (int msk = 1; msk < 16; msk <<= 1) {
#pragma unroll
      for (int r = 0; r < 8; ++r) {
        sa[r] += __shfl_xor(sa[r], msk, 16);
        sb[r] += __shfl_xor(sb[r], msk, 16);
      }
    }

    if (m == 0) {
#pragma unroll
      for (int r = 0; r < 8; ++r) {
        int o0 = tbase + half * 8 + r;
        int o1 = tbase + 16 + half * 8 + r;
        if (o0 < n_edges) out[o0] = sa[r] + b2v;
        if (o1 < n_edges) out[o1] = sb[r] + b2v;
      }
    }
  }
}

extern "C" void kernel_launch(void* const* d_in, const int* in_sizes, int n_in,
                              void* d_out, int out_size, void* d_ws, size_t ws_size,
                              hipStream_t stream) {
  const float*     src = (const float*)d_in[0];
  const float*     dst = (const float*)d_in[1];
  const long long* eli = (const long long*)d_in[2];  // edge_label_index, int64 [2,E]
  const float*     W1  = (const float*)d_in[3];
  const float*     b1  = (const float*)d_in[4];
  const float*     W2  = (const float*)d_in[5];
  const float*     b2  = (const float*)d_in[6];
  float*           out = (float*)d_out;

  const int n_edges     = in_sizes[2] / 2;
  const int total_tiles = (n_edges + 15) / 16;
  const int tiles_blk   = WAVES * TPP * 2;
  const int grid        = (total_tiles + tiles_blk - 1) / tiles_blk;

  edge_decoder_kernel<<<grid, 256, 0, stream>>>(src, dst, eli, W1, b1, W2, b2,
                                                out, n_edges);
}
